// MixtralDecoderLayer_59047210385669
// MI455X (gfx1250) — matrix operand order repaired and checked by reading, not compiled, not run on hardware
//
#include <hip/hip_runtime.h>
#include <hip/hip_bf16.h>
#include <string.h>
#include <stdint.h>

// ---------------------------------------------------------------------------
// Mixtral decoder layer for MI455X (gfx1250, wave32, WMMA bf16 -> f32 accum)
// ---------------------------------------------------------------------------

#define HDIM_   64
#define NH_     16
#define NKV_    8
#define HID_    1024
#define SEQ_    2048
#define BATCH_  2
#define TOK_    4096          // BATCH_*SEQ_
#define EXP_    8
#define IDIM_   3584

typedef __attribute__((ext_vector_type(16))) __bf16 v16bf;
typedef __attribute__((ext_vector_type(8)))  float  v8f;

// ---------------- WMMA helpers (m/n per CDNA5 ISA 7.12.2, k = 16*half + j) ---------------

__device__ __forceinline__ v8f wmma_bf16(v16bf a, v16bf b, v8f c) {
  return __builtin_amdgcn_wmma_f32_16x16x32_bf16(
      false, a, false, b, (short)0, c, false, false);
}

// A fragment: tile [16 rows M][>=32 cols K] row-major, rows 16B-aligned.
__device__ __forceinline__ v16bf load_frag_a(const __bf16* tile, int stride, int kbase) {
  const int lane = threadIdx.x & 31;
  const int half = lane >> 4;
  const __bf16* r = tile + (size_t)(lane & 15) * stride + kbase + half * 16;
  v16bf a;
#pragma unroll
  for (int j = 0; j < 16; ++j) a[j] = r[j];
  return a;
}

// B fragment from a transposed [N][K] tile (rows 16B-aligned).
__device__ __forceinline__ v16bf load_frag_bT(const __bf16* tile, int stride, int kbase) {
  const int lane = threadIdx.x & 31;
  const int half = lane >> 4;
  const __bf16* p = tile + (size_t)(lane & 15) * stride + kbase + half * 16;
  v16bf b;
#pragma unroll
  for (int j = 0; j < 16; ++j) b[j] = p[j];
  return b;
}

// ---------------- CDNA5 async global->LDS (ISA 15.18.3, ASYNCcnt) ----------

__device__ __forceinline__ void async_load_b128(const __bf16* gsrc, __bf16* ldst) {
  const uint32_t loff = (uint32_t)(uintptr_t)ldst;
  const uint64_t ga   = (uint64_t)(uintptr_t)gsrc;
  asm volatile("global_load_async_to_lds_b128 %0, %1, off"
               :: "v"(loff), "v"(ga) : "memory");
}
__device__ __forceinline__ void wait_async0() {
  asm volatile("s_wait_asynccnt 0x0" ::: "memory");
}

// ---------------- small utility kernels -------------------------------------

__global__ void zero_f32_kernel(float* p, int n) {
  for (int i = blockIdx.x * blockDim.x + threadIdx.x; i < n; i += gridDim.x * blockDim.x)
    p[i] = 0.0f;
}
__global__ void zero_i32_kernel(int* p, int n) {
  int i = blockIdx.x * blockDim.x + threadIdx.x;
  if (i < n) p[i] = 0;
}
__global__ void cvt_f32_bf16_kernel(const float* __restrict__ s, __bf16* __restrict__ d, int n) {
  for (int i = blockIdx.x * blockDim.x + threadIdx.x; i < n; i += gridDim.x * blockDim.x)
    d[i] = (__bf16)s[i];
}
__global__ void final_add_kernel(const float* __restrict__ a, const float* __restrict__ b,
                                 float* __restrict__ o, int n) {
  for (int i = blockIdx.x * blockDim.x + threadIdx.x; i < n; i += gridDim.x * blockDim.x)
    o[i] = a[i] + b[i];
}

// ---------------- RMSNorm (fp32 in -> bf16 out) -----------------------------

__global__ void rmsnorm_kernel(const float* __restrict__ x, const float* __restrict__ w,
                               __bf16* __restrict__ out) {
  __shared__ float red[256];
  const int t = blockIdx.x;
  const float* xr = x + (size_t)t * HID_;
  float ss = 0.0f;
  for (int i = threadIdx.x; i < HID_; i += 256) { float v = xr[i]; ss += v * v; }
  red[threadIdx.x] = ss;
  __syncthreads();
  for (int st = 128; st > 0; st >>= 1) {
    if (threadIdx.x < st) red[threadIdx.x] += red[threadIdx.x + st];
    __syncthreads();
  }
  const float scale = rsqrtf(red[0] * (1.0f / (float)HID_) + 1e-6f);
  for (int i = threadIdx.x; i < HID_; i += 256)
    out[(size_t)t * HID_ + i] = (__bf16)(xr[i] * scale * w[i]);
}

// ---------------- generic bf16 GEMM: C[M,N] = A[M,K] x B[K,N] ---------------
// 256 thr = 8 waves as 4(M) x 2(N); block tile 128x64; wave tile 32x32.
// Double-buffered LDS: async A-DMA + register-held B for tile t+1 issued
// before the WMMAs of tile t; transpose-store + asynccnt wait after.
// MODE 0: Cf = acc ; MODE 1: Cf = resid + acc

template <int MODE>
__global__ void gemm_bf16_kernel(const __bf16* __restrict__ A, const __bf16* __restrict__ B,
                                 float* __restrict__ Cf, const float* __restrict__ resid,
                                 int M, int N, int K) {
  __shared__ __bf16 As[2][128][40];     // [buf][m][k]
  __shared__ __bf16 BsT[2][64][40];     // [buf][n][k]
  const int tid  = threadIdx.x;
  const int wave = tid >> 5;
  const int wm   = wave >> 1;
  const int wn   = wave & 1;
  const int m0   = blockIdx.y * 128;
  const int n0   = blockIdx.x * 64;
  v8f acc[2][2] = {{{}, {}}, {{}, {}}};

  const int ar = tid >> 1;             // A: row 0..127
  const int ac = (tid & 1) * 16;       // A: col 0 or 16
  const int br = tid >> 3;             // B: k-row 0..31
  const int bc = (tid & 7) * 8;        // B: n-col 0..56
  const __bf16* arow = A + (size_t)(m0 + ar) * K + ac;

  // prologue: stage tile 0 into buffer 0
  async_load_b128(arow,     &As[0][ar][ac]);
  async_load_b128(arow + 8, &As[0][ar][ac + 8]);
  {
    const __bf16* srcB = B + (size_t)br * N + n0 + bc;
#pragma unroll
    for (int j = 0; j < 8; ++j) BsT[0][bc + j][br] = srcB[j];
  }
  wait_async0();
  __syncthreads();

  const int NK = K >> 5;
  for (int t = 0; t < NK; ++t) {
    const int cur = t & 1, nxt = cur ^ 1;
    __bf16 breg[8];
    if (t + 1 < NK) {                                // issue next-tile traffic early
      const int k0 = (t + 1) << 5;
      async_load_b128(arow + k0,     &As[nxt][ar][ac]);
      async_load_b128(arow + k0 + 8, &As[nxt][ar][ac + 8]);
      const __bf16* srcB = B + (size_t)(k0 + br) * N + n0 + bc;
#pragma unroll
      for (int j = 0; j < 8; ++j) breg[j] = srcB[j];
    }
    v16bf a0 = load_frag_a(&As[cur][wm * 32][0], 40, 0);
    v16bf a1 = load_frag_a(&As[cur][wm * 32 + 16][0], 40, 0);
    v16bf b0 = load_frag_bT(&BsT[cur][wn * 32][0], 40, 0);
    v16bf b1 = load_frag_bT(&BsT[cur][wn * 32 + 16][0], 40, 0);
    acc[0][0] = wmma_bf16(a0, b0, acc[0][0]);
    acc[0][1] = wmma_bf16(a0, b1, acc[0][1]);
    acc[1][0] = wmma_bf16(a1, b0, acc[1][0]);
    acc[1][1] = wmma_bf16(a1, b1, acc[1][1]);
    if (t + 1 < NK) {
#pragma unroll
      for (int j = 0; j < 8; ++j) BsT[nxt][bc + j][br] = breg[j];
      wait_async0();
    }
    __syncthreads();
  }

  const int lane = tid & 31, half = lane >> 4, nn = lane & 15;
#pragma unroll
  for (int rm = 0; rm < 2; ++rm) {
    const int row0 = m0 + wm * 32 + rm * 16 + half * 8;
#pragma unroll
    for (int ns = 0; ns < 2; ++ns) {
      const int col = n0 + wn * 32 + ns * 16 + nn;
      float* cp = Cf + (size_t)row0 * N + col;
      const float* rp = resid + (size_t)row0 * N + col;
#pragma unroll
      for (int r = 0; r < 8; ++r) {
        if (MODE == 1) cp[0] = rp[0] + acc[rm][ns][r];
        else           cp[0] = acc[rm][ns][r];
        cp += N; rp += N;
      }
    }
  }
}

// ---------------- RoPE + bf16 pack for q/k/v --------------------------------

__global__ void rope_kernel(const float* __restrict__ qf, const float* __restrict__ kf,
                            const float* __restrict__ vf, const int* __restrict__ pos_ids,
                            const float* __restrict__ rsin, const float* __restrict__ rcos,
                            __bf16* __restrict__ qb, __bf16* __restrict__ kb,
                            __bf16* __restrict__ vb) {
  const int t = blockIdx.x;
  const int pos = pos_ids[t];
  for (int d = threadIdx.x; d < 2048; d += 256) {
    if (d < 1024) {                                // q
      const int dd = d & 63;
      const float c = rcos[pos * 64 + dd], s = rsin[pos * 64 + dd];
      const size_t base = (size_t)t * 1024;
      const float x = qf[base + d];
      const float rot = (dd < 32) ? -qf[base + d + 32] : qf[base + d - 32];
      qb[base + d] = (__bf16)(x * c + rot * s);
    } else if (d < 1536) {                         // k
      const int dk = d - 1024;
      const int dd = dk & 63;
      const float c = rcos[pos * 64 + dd], s = rsin[pos * 64 + dd];
      const size_t base = (size_t)t * 512;
      const float x = kf[base + dk];
      const float rot = (dd < 32) ? -kf[base + dk + 32] : kf[base + dk - 32];
      kb[base + dk] = (__bf16)(x * c + rot * s);
    } else {                                       // v
      const int dv = d - 1536;
      vb[(size_t)t * 512 + dv] = (__bf16)vf[(size_t)t * 512 + dv];
    }
  }
}

// ---------------- flash attention (WMMA QK^T and P.V, online softmax) -------
// grid: (S/64, NH, B), block 128 (4 waves x 16 q-rows). GQA: kvh = h>>1.
// Double-buffered K/V tiles; kv columns permuted in-tile so each lane's
// probability pair packs into one 32-bit LDS store.

__global__ void attention_kernel(const __bf16* __restrict__ q, const __bf16* __restrict__ k,
                                 const __bf16* __restrict__ v, const int* __restrict__ amask,
                                 __bf16* __restrict__ ctx) {
  __shared__ __bf16 Kt[2][32][72];        // [buf][key][hd]
  __shared__ __bf16 VtT[2][64][40];       // [buf][hd][kvcol]
  __shared__ unsigned int Plu[4][16][20]; // per-wave P, packed bf16 pairs
  const int qtile = blockIdx.x;
  const int h     = blockIdx.y;
  const int bb    = blockIdx.z;
  const int kvh   = h >> 1;
  const int tid   = threadIdx.x;
  const int w     = tid >> 5;
  const int lane  = tid & 31;
  const int half  = lane >> 4;
  const int nn    = lane & 15;
  const int qbase = qtile * 64;
  const int qrow0 = qbase + w * 16;

  const __bf16* qptr = q + (size_t)(bb * SEQ_ + qrow0) * (NH_ * HDIM_) + h * HDIM_;
  v16bf qa0 = load_frag_a(qptr, NH_ * HDIM_, 0);
  v16bf qa1 = load_frag_a(qptr, NH_ * HDIM_, 32);

  v8f acc[4]; float mst[8], lst[8];
#pragma unroll
  for (int nt = 0; nt < 4; ++nt) acc[nt] = (v8f){};
#pragma unroll
  for (int r = 0; r < 8; ++r) { mst[r] = -1e30f; lst[r] = 0.0f; }

  const int key   = tid >> 2;
  const int c16   = (tid & 3) * 16;
  const int kvcol = 2 * (key & 15) + (key >> 4);
  const __bf16* kbase = k + (size_t)(bb * SEQ_ + key) * (NKV_ * HDIM_) + kvh * HDIM_ + c16;
  const __bf16* vbase = v + (size_t)(bb * SEQ_ + key) * (NKV_ * HDIM_) + kvh * HDIM_ + c16;
  const size_t  jstep = (size_t)32 * (NKV_ * HDIM_);

  // prologue: stage kv tile 0 into buffer 0
  async_load_b128(kbase, &Kt[0][key][c16]);
#pragma unroll
  for (int j = 0; j < 16; ++j) VtT[0][c16 + j][kvcol] = vbase[j];
  wait_async0();
  __syncthreads();

  const int ntiles = (qbase + 64) >> 5;            // causal bound
  for (int t = 0; t < ntiles; ++t) {
    const int cur = t & 1, nxt = cur ^ 1;
    __bf16 vreg[16];
    if (t + 1 < ntiles) {                          // issue next-tile traffic early
      async_load_b128(kbase + (size_t)(t + 1) * jstep, &Kt[nxt][key][c16]);
      const __bf16* vs = vbase + (size_t)(t + 1) * jstep;
#pragma unroll
      for (int j = 0; j < 16; ++j) vreg[j] = vs[j];
    }
    const int j0 = t * 32;

    float sc[2][8];
#pragma unroll
    for (int nt = 0; nt < 2; ++nt) {
      v8f s = {};
      v16bf kb0 = load_frag_bT(&Kt[cur][nt * 16][0], 72, 0);
      v16bf kb1 = load_frag_bT(&Kt[cur][nt * 16][0], 72, 32);
      s = wmma_bf16(qa0, kb0, s);
      s = wmma_bf16(qa1, kb1, s);
      const int colg = j0 + nt * 16 + nn;
      const int mok  = amask[bb * SEQ_ + colg];
#pragma unroll
      for (int r = 0; r < 8; ++r) {
        const int rowg = qrow0 + half * 8 + r;
        float sv = s[r] * 0.125f;                  // 1/sqrt(64)
        if (colg > rowg || mok <= 0) sv = -1e30f;
        sc[nt][r] = sv;
      }
    }

    float mt[8];
#pragma unroll
    for (int r = 0; r < 8; ++r) mt[r] = fmaxf(sc[0][r], sc[1][r]);
#pragma unroll
    for (int r = 0; r < 8; ++r) {
      mt[r] = fmaxf(mt[r], __shfl_xor(mt[r], 1, 32));
      mt[r] = fmaxf(mt[r], __shfl_xor(mt[r], 2, 32));
      mt[r] = fmaxf(mt[r], __shfl_xor(mt[r], 4, 32));
      mt[r] = fmaxf(mt[r], __shfl_xor(mt[r], 8, 32));
    }

    float rs[8], scl[8];
#pragma unroll
    for (int r = 0; r < 8; ++r) {
      const float mnew = fmaxf(mst[r], mt[r]);
      scl[r] = __expf(mst[r] - mnew);
      const float p0 = __expf(sc[0][r] - mnew);
      const float p1 = __expf(sc[1][r] - mnew);
      mst[r] = mnew;
      rs[r] = p0 + p1;
      const __bf16 b0 = (__bf16)p0, b1 = (__bf16)p1;
      unsigned short u0, u1;
      memcpy(&u0, &b0, 2); memcpy(&u1, &b1, 2);
      Plu[w][half * 8 + r][nn] = (unsigned)u0 | ((unsigned)u1 << 16);
#pragma unroll
      for (int nt = 0; nt < 4; ++nt) acc[nt][r] *= scl[r];
    }
#pragma unroll
    for (int r = 0; r < 8; ++r) {
      rs[r] += __shfl_xor(rs[r], 1, 32);
      rs[r] += __shfl_xor(rs[r], 2, 32);
      rs[r] += __shfl_xor(rs[r], 4, 32);
      rs[r] += __shfl_xor(rs[r], 8, 32);
      lst[r] = lst[r] * scl[r] + rs[r];
    }

    __syncthreads();                               // P visible
    v16bf pa = load_frag_a((const __bf16*)&Plu[w][0][0], 40, 0);
#pragma unroll
    for (int nt = 0; nt < 4; ++nt) {
      v16bf vb = load_frag_bT(&VtT[cur][nt * 16][0], 40, 0);
      acc[nt] = wmma_bf16(pa, vb, acc[nt]);
    }
    if (t + 1 < ntiles) {
#pragma unroll
      for (int j = 0; j < 16; ++j) VtT[nxt][c16 + j][kvcol] = vreg[j];
      wait_async0();
    }
    __syncthreads();                               // cur reads done everywhere
  }

#pragma unroll
  for (int nt = 0; nt < 4; ++nt) {
#pragma unroll
    for (int r = 0; r < 8; ++r) {
      const int rowg = qrow0 + half * 8 + r;
      const float val = acc[nt][r] / lst[r];
      ctx[(size_t)(bb * SEQ_ + rowg) * (NH_ * HDIM_) + h * HDIM_ + nt * 16 + nn] = (__bf16)val;
    }
  }
}

// ---------------- router: logits, softmax, top-2, compaction ----------------

__global__ void router_kernel(const __bf16* __restrict__ h2, const float* __restrict__ gate,
                              int* __restrict__ counts, int* __restrict__ tokens,
                              float* __restrict__ wts) {
  const int lane = threadIdx.x & 31;
  const int w    = threadIdx.x >> 5;
  const int t    = blockIdx.x * 8 + w;
  float acc[EXP_];
#pragma unroll
  for (int e = 0; e < EXP_; ++e) acc[e] = 0.0f;
  for (int s = 0; s < 32; ++s) {
    const int i = s * 32 + lane;
    const float x = (float)h2[(size_t)t * HID_ + i];
#pragma unroll
    for (int e = 0; e < EXP_; ++e) acc[e] += x * gate[i * EXP_ + e];
  }
#pragma unroll
  for (int e = 0; e < EXP_; ++e) {
    acc[e] += __shfl_xor(acc[e], 16, 32);
    acc[e] += __shfl_xor(acc[e], 8, 32);
    acc[e] += __shfl_xor(acc[e], 4, 32);
    acc[e] += __shfl_xor(acc[e], 2, 32);
    acc[e] += __shfl_xor(acc[e], 1, 32);
  }
  if (lane == 0) {
    float m = acc[0];
#pragma unroll
    for (int e = 1; e < EXP_; ++e) m = fmaxf(m, acc[e]);
    float p[EXP_];
#pragma unroll
    for (int e = 0; e < EXP_; ++e) p[e] = __expf(acc[e] - m);
    int a = 0;
#pragma unroll
    for (int e = 1; e < EXP_; ++e) if (p[e] > p[a]) a = e;
    int b = (a == 0) ? 1 : 0;
#pragma unroll
    for (int e = 0; e < EXP_; ++e) if (e != a && p[e] > p[b]) b = e;
    const float inv = 1.0f / (p[a] + p[b]);
    const int sa = atomicAdd(&counts[a], 1);
    tokens[a * TOK_ + sa] = t; wts[a * TOK_ + sa] = p[a] * inv;
    const int sb = atomicAdd(&counts[b], 1);
    tokens[b * TOK_ + sb] = t; wts[b * TOK_ + sb] = p[b] * inv;
  }
}

// ---------------- MoE up: u = silu(h.w1) * (h.w3) (gathered rows) -----------
// block tile 128(slots) x 64(I-cols); 8 WMMA per wave per k-step; dbl-buffered.

__global__ void moe_up_kernel(const __bf16* __restrict__ h2, const __bf16* __restrict__ w1e,
                              const __bf16* __restrict__ w3e, const int* __restrict__ tokens,
                              const int* __restrict__ cnt, __bf16* __restrict__ u) {
  const int count = *cnt;
  const int m0 = blockIdx.y * 128;
  if (m0 >= count) return;
  __shared__ __bf16 As[2][128][40];
  __shared__ __bf16 B1T[2][64][40];
  __shared__ __bf16 B3T[2][64][40];
  const int tid  = threadIdx.x;
  const int wave = tid >> 5, wm = wave >> 1, wn = wave & 1;
  const int n0   = blockIdx.x * 64;
  v8f ag[2][2] = {{{}, {}}, {{}, {}}};
  v8f at[2][2] = {{{}, {}}, {{}, {}}};

  const int ar = tid >> 1;
  const int ac = (tid & 1) * 16;
  const int slot = m0 + ar;
  const int cs = (slot < count) ? slot : (count - 1);   // clamp, no value select
  const int trow = tokens[cs];
  const int br = tid >> 3;
  const int bc = (tid & 7) * 8;
  const __bf16* arow = h2 + (size_t)trow * HID_ + ac;

  async_load_b128(arow,     &As[0][ar][ac]);
  async_load_b128(arow + 8, &As[0][ar][ac + 8]);
  {
    const __bf16* s1 = w1e + (size_t)br * IDIM_ + n0 + bc;
    const __bf16* s3 = w3e + (size_t)br * IDIM_ + n0 + bc;
#pragma unroll
    for (int j = 0; j < 8; ++j) { B1T[0][bc + j][br] = s1[j]; B3T[0][bc + j][br] = s3[j]; }
  }
  wait_async0();
  __syncthreads();

  const int NK = HID_ >> 5;
  for (int t = 0; t < NK; ++t) {
    const int cur = t & 1, nxt = cur ^ 1;
    __bf16 r1[8], r3[8];
    if (t + 1 < NK) {
      const int k0 = (t + 1) << 5;
      async_load_b128(arow + k0,     &As[nxt][ar][ac]);
      async_load_b128(arow + k0 + 8, &As[nxt][ar][ac + 8]);
      const __bf16* s1 = w1e + (size_t)(k0 + br) * IDIM_ + n0 + bc;
      const __bf16* s3 = w3e + (size_t)(k0 + br) * IDIM_ + n0 + bc;
#pragma unroll
      for (int j = 0; j < 8; ++j) { r1[j] = s1[j]; r3[j] = s3[j]; }
    }
    v16bf a0  = load_frag_a(&As[cur][wm * 32][0], 40, 0);
    v16bf a1  = load_frag_a(&As[cur][wm * 32 + 16][0], 40, 0);
    v16bf b10 = load_frag_bT(&B1T[cur][wn * 32][0], 40, 0);
    v16bf b11 = load_frag_bT(&B1T[cur][wn * 32 + 16][0], 40, 0);
    v16bf b30 = load_frag_bT(&B3T[cur][wn * 32][0], 40, 0);
    v16bf b31 = load_frag_bT(&B3T[cur][wn * 32 + 16][0], 40, 0);
    ag[0][0] = wmma_bf16(a0, b10, ag[0][0]);
    ag[0][1] = wmma_bf16(a0, b11, ag[0][1]);
    ag[1][0] = wmma_bf16(a1, b10, ag[1][0]);
    ag[1][1] = wmma_bf16(a1, b11, ag[1][1]);
    at[0][0] = wmma_bf16(a0, b30, at[0][0]);
    at[0][1] = wmma_bf16(a0, b31, at[0][1]);
    at[1][0] = wmma_bf16(a1, b30, at[1][0]);
    at[1][1] = wmma_bf16(a1, b31, at[1][1]);
    if (t + 1 < NK) {
#pragma unroll
      for (int j = 0; j < 8; ++j) { B1T[nxt][bc + j][br] = r1[j]; B3T[nxt][bc + j][br] = r3[j]; }
      wait_async0();
    }
    __syncthreads();
  }

  const int lane = tid & 31, half = lane >> 4, nn = lane & 15;
#pragma unroll
  for (int rm = 0; rm < 2; ++rm) {
    const int sl0 = m0 + wm * 32 + rm * 16 + half * 8;
#pragma unroll
    for (int ns = 0; ns < 2; ++ns) {
      const int col = n0 + wn * 32 + ns * 16 + nn;
#pragma unroll
      for (int r = 0; r < 8; ++r) {
        const int sl = sl0 + r;
        if (sl < count) {
          const float g = ag[rm][ns][r];
          const float act = g / (1.0f + __expf(-g));
          u[(size_t)sl * IDIM_ + col] = (__bf16)(act * at[rm][ns][r]);
        }
      }
    }
  }
}

// ---------------- MoE down: moe_out[t] += w * (u . w2) ----------------------

__global__ void moe_down_kernel(const __bf16* __restrict__ u, const __bf16* __restrict__ w2e,
                                const int* __restrict__ tokens, const float* __restrict__ wts,
                                const int* __restrict__ cnt, float* __restrict__ moe_out) {
  const int count = *cnt;
  const int m0 = blockIdx.y * 128;
  if (m0 >= count) return;
  __shared__ __bf16 As[2][128][40];
  __shared__ __bf16 BsT[2][64][40];
  const int tid  = threadIdx.x;
  const int wave = tid >> 5, wm = wave >> 1, wn = wave & 1;
  const int n0   = blockIdx.x * 64;
  v8f acc[2][2] = {{{}, {}}, {{}, {}}};

  const int ar = tid >> 1;
  const int ac = (tid & 1) * 16;
  const int slot = m0 + ar;
  const int cs = (slot < count) ? slot : (count - 1);
  const int br = tid >> 3;
  const int bc = (tid & 7) * 8;
  const __bf16* arow = u + (size_t)cs * IDIM_ + ac;

  async_load_b128(arow,     &As[0][ar][ac]);
  async_load_b128(arow + 8, &As[0][ar][ac + 8]);
  {
    const __bf16* srcB = w2e + (size_t)br * HID_ + n0 + bc;
#pragma unroll
    for (int j = 0; j < 8; ++j) BsT[0][bc + j][br] = srcB[j];
  }
  wait_async0();
  __syncthreads();

  const int NK = IDIM_ >> 5;
  for (int t = 0; t < NK; ++t) {
    const int cur = t & 1, nxt = cur ^ 1;
    __bf16 breg[8];
    if (t + 1 < NK) {
      const int k0 = (t + 1) << 5;
      async_load_b128(arow + k0,     &As[nxt][ar][ac]);
      async_load_b128(arow + k0 + 8, &As[nxt][ar][ac + 8]);
      const __bf16* srcB = w2e + (size_t)(k0 + br) * HID_ + n0 + bc;
#pragma unroll
      for (int j = 0; j < 8; ++j) breg[j] = srcB[j];
    }
    v16bf a0 = load_frag_a(&As[cur][wm * 32][0], 40, 0);
    v16bf a1 = load_frag_a(&As[cur][wm * 32 + 16][0], 40, 0);
    v16bf b0 = load_frag_bT(&BsT[cur][wn * 32][0], 40, 0);
    v16bf b1 = load_frag_bT(&BsT[cur][wn * 32 + 16][0], 40, 0);
    acc[0][0] = wmma_bf16(a0, b0, acc[0][0]);
    acc[0][1] = wmma_bf16(a0, b1, acc[0][1]);
    acc[1][0] = wmma_bf16(a1, b0, acc[1][0]);
    acc[1][1] = wmma_bf16(a1, b1, acc[1][1]);
    if (t + 1 < NK) {
#pragma unroll
      for (int j = 0; j < 8; ++j) BsT[nxt][bc + j][br] = breg[j];
      wait_async0();
    }
    __syncthreads();
  }

  const int lane = tid & 31, half = lane >> 4, nn = lane & 15;
#pragma unroll
  for (int rm = 0; rm < 2; ++rm) {
    const int sl0 = m0 + wm * 32 + rm * 16 + half * 8;
#pragma unroll
    for (int ns = 0; ns < 2; ++ns) {
      const int col = n0 + wn * 32 + ns * 16 + nn;
#pragma unroll
      for (int r = 0; r < 8; ++r) {
        const int sl = sl0 + r;
        if (sl < count) {
          const int t = tokens[sl];
          moe_out[(size_t)t * HID_ + col] += wts[sl] * acc[rm][ns][r];
        }
      }
    }
  }
}

// ---------------------------------------------------------------------------
// host-side orchestration
// ---------------------------------------------------------------------------

extern "C" void kernel_launch(void* const* d_in, const int* in_sizes, int n_in,
                              void* d_out, int out_size, void* d_ws, size_t ws_size,
                              hipStream_t stream) {
  (void)in_sizes; (void)n_in; (void)out_size; (void)ws_size;
  const float* x      = (const float*)d_in[0];
  const int*   amask  = (const int*)d_in[1];
  const int*   posids = (const int*)d_in[2];
  const float* rsin   = (const float*)d_in[3];
  const float* rcos   = (const float*)d_in[4];
  const float* ln1    = (const float*)d_in[5];
  const float* ln2    = (const float*)d_in[6];
  const float* wq     = (const float*)d_in[7];
  const float* wk     = (const float*)d_in[8];
  const float* wv     = (const float*)d_in[9];
  const float* wo     = (const float*)d_in[10];
  const float* gate   = (const float*)d_in[11];
  const float* w1     = (const float*)d_in[12];
  const float* w2     = (const float*)d_in[13];
  const float* w3     = (const float*)d_in[14];
  float* out = (float*)d_out;

  char* ws = (char*)d_ws;
  size_t off = 0;
  auto alloc = [&](size_t bytes) -> char* {
    char* p = ws + off;
    off += (bytes + 255) & ~(size_t)255;
    return p;
  };

  __bf16* wq_b  = (__bf16*)alloc((size_t)HID_ * 1024 * 2);
  __bf16* wk_b  = (__bf16*)alloc((size_t)HID_ * 512 * 2);
  __bf16* wv_b  = (__bf16*)alloc((size_t)HID_ * 512 * 2);
  __bf16* wo_b  = (__bf16*)alloc((size_t)1024 * HID_ * 2);
  __bf16* w1_b  = (__bf16*)alloc((size_t)EXP_ * HID_ * IDIM_ * 2);
  __bf16* w2_b  = (__bf16*)alloc((size_t)EXP_ * IDIM_ * HID_ * 2);
  __bf16* w3_b  = (__bf16*)alloc((size_t)EXP_ * HID_ * IDIM_ * 2);
  __bf16* h1_b  = (__bf16*)alloc((size_t)TOK_ * HID_ * 2);
  float*  qf    = (float*) alloc((size_t)TOK_ * 1024 * 4);
  float*  kf    = (float*) alloc((size_t)TOK_ * 512 * 4);
  float*  vf    = (float*) alloc((size_t)TOK_ * 512 * 4);
  __bf16* qb    = (__bf16*)alloc((size_t)TOK_ * 1024 * 2);
  __bf16* kb    = (__bf16*)alloc((size_t)TOK_ * 512 * 2);
  __bf16* vb    = (__bf16*)alloc((size_t)TOK_ * 512 * 2);
  __bf16* ctx_b = (__bf16*)alloc((size_t)TOK_ * 1024 * 2);
  float*  x1    = (float*) alloc((size_t)TOK_ * HID_ * 4);
  __bf16* h2_b  = (__bf16*)alloc((size_t)TOK_ * HID_ * 2);
  float*  moeo  = (float*) alloc((size_t)TOK_ * HID_ * 4);
  __bf16* u_b   = (__bf16*)alloc((size_t)TOK_ * IDIM_ * 2);
  int*    cnts  = (int*)   alloc(EXP_ * 4);
  int*    toks  = (int*)   alloc((size_t)EXP_ * TOK_ * 4);
  float*  twts  = (float*) alloc((size_t)EXP_ * TOK_ * 4);

  // 0) clear routing counters + MoE accumulator
  zero_i32_kernel<<<1, 32, 0, stream>>>(cnts, EXP_);
  zero_f32_kernel<<<2048, 256, 0, stream>>>(moeo, TOK_ * HID_);

  // 1) weights fp32 -> bf16
  cvt_f32_bf16_kernel<<<2048, 256, 0, stream>>>(wq, wq_b, HID_ * 1024);
  cvt_f32_bf16_kernel<<<2048, 256, 0, stream>>>(wk, wk_b, HID_ * 512);
  cvt_f32_bf16_kernel<<<2048, 256, 0, stream>>>(wv, wv_b, HID_ * 512);
  cvt_f32_bf16_kernel<<<2048, 256, 0, stream>>>(wo, wo_b, 1024 * HID_);
  cvt_f32_bf16_kernel<<<8192, 256, 0, stream>>>(w1, w1_b, EXP_ * HID_ * IDIM_);
  cvt_f32_bf16_kernel<<<8192, 256, 0, stream>>>(w2, w2_b, EXP_ * IDIM_ * HID_);
  cvt_f32_bf16_kernel<<<8192, 256, 0, stream>>>(w3, w3_b, EXP_ * HID_ * IDIM_);

  // 2) pre-attention RMSNorm
  rmsnorm_kernel<<<TOK_, 256, 0, stream>>>(x, ln1, h1_b);

  // 3) QKV projections (WMMA)
  gemm_bf16_kernel<0><<<dim3(1024 / 64, TOK_ / 128), 256, 0, stream>>>(
      h1_b, wq_b, qf, nullptr, TOK_, 1024, HID_);
  gemm_bf16_kernel<0><<<dim3(512 / 64, TOK_ / 128), 256, 0, stream>>>(
      h1_b, wk_b, kf, nullptr, TOK_, 512, HID_);
  gemm_bf16_kernel<0><<<dim3(512 / 64, TOK_ / 128), 256, 0, stream>>>(
      h1_b, wv_b, vf, nullptr, TOK_, 512, HID_);

  // 4) RoPE + bf16 pack
  rope_kernel<<<TOK_, 256, 0, stream>>>(qf, kf, vf, posids, rsin, rcos, qb, kb, vb);

  // 5) flash attention (WMMA)
  attention_kernel<<<dim3(SEQ_ / 64, NH_, BATCH_), 128, 0, stream>>>(qb, kb, vb, amask, ctx_b);

  // 6) O projection + residual
  gemm_bf16_kernel<1><<<dim3(HID_ / 64, TOK_ / 128), 256, 0, stream>>>(
      ctx_b, wo_b, x1, x, TOK_, HID_, 1024);

  // 7) pre-MoE RMSNorm + routing
  rmsnorm_kernel<<<TOK_, 256, 0, stream>>>(x1, ln2, h2_b);
  router_kernel<<<TOK_ / 8, 256, 0, stream>>>(h2_b, gate, cnts, toks, twts);

  // 8) routed experts (sequential on stream; u buffer reused)
  for (int e = 0; e < EXP_; ++e) {
    moe_up_kernel<<<dim3(IDIM_ / 64, TOK_ / 128), 256, 0, stream>>>(
        h2_b, w1_b + (size_t)e * HID_ * IDIM_, w3_b + (size_t)e * HID_ * IDIM_,
        toks + (size_t)e * TOK_, cnts + e, u_b);
    moe_down_kernel<<<dim3(HID_ / 64, TOK_ / 128), 256, 0, stream>>>(
        u_b, w2_b + (size_t)e * IDIM_ * HID_,
        toks + (size_t)e * TOK_, twts + (size_t)e * TOK_, cnts + e, moeo);
  }

  // 9) final residual
  final_add_kernel<<<2048, 256, 0, stream>>>(x1, moeo, out, TOK_ * HID_);
}
